// MultiHeadAttention_8761733284123
// MI455X (gfx1250) — matrix operand order repaired and checked by reading, not compiled, not run on hardware
//
#include <hip/hip_runtime.h>
#include <hip/hip_bf16.h>

// ---------------------------------------------------------------------------
// MI455X (gfx1250) causal multi-head attention forward.
// wave32, v_wmma_f32_16x16x32_bf16, flash-style online softmax.
// Projection GEMMs: fully async double-buffered A+B tile staging in LDS
// (ASYNCcnt-only main loop; no LOADcnt stalls at barriers).
// ---------------------------------------------------------------------------

typedef __bf16 bf16_t;
typedef __bf16 v16bf __attribute__((ext_vector_type(16)));
typedef __bf16 v8bf  __attribute__((ext_vector_type(8)));
typedef float  v8f   __attribute__((ext_vector_type(8)));

#define B_   2
#define S_   2048
#define E_   1024
#define H_   16
#define D_   64

#define INFF __builtin_inff()

__device__ __forceinline__ v8f wmma_bf16(v16bf a, v16bf b, v8f c) {
  // 8 args: (neg_a, A, neg_b, B, c_mod, C, reuse_a, reuse_b)
  return __builtin_amdgcn_wmma_f32_16x16x32_bf16(false, a, false, b, (short)0, c,
                                                 false, false);
}

// Load a 16-half fragment from two 16B chunks.
__device__ __forceinline__ v16bf ld16(const bf16_t* p0, const bf16_t* p1) {
  v8bf lo = *(const v8bf*)p0;
  v8bf hi = *(const v8bf*)p1;
  return __builtin_shufflevector(lo, hi, 0, 1, 2, 3, 4, 5, 6, 7,
                                 8, 9, 10, 11, 12, 13, 14, 15);
}

__device__ __forceinline__ v8f vzero8() {
  v8f z = {0.f, 0.f, 0.f, 0.f, 0.f, 0.f, 0.f, 0.f};
  return z;
}

// Async global -> LDS 16B copy (ASYNCcnt-tracked).
__device__ __forceinline__ void async_copy_b128(unsigned lds_off, const bf16_t* gptr) {
  asm volatile("global_load_async_to_lds_b128 %0, %1, off"
               :: "v"(lds_off), "v"((unsigned long long)(uintptr_t)gptr)
               : "memory");
}
// Async loads complete in order: waiting <=4 retires the 4 ops of the tile
// issued one step earlier while the 4 just-issued remain in flight.
__device__ __forceinline__ void wait_async_le4() {
  asm volatile("s_wait_asynccnt 0x4" ::: "memory");
}

// ---------------------------------------------------------------------------
// fp32 -> bf16 elementwise convert
// ---------------------------------------------------------------------------
__global__ void f32_to_bf16_kernel(const float* __restrict__ in,
                                   bf16_t* __restrict__ out, int n) {
  int i = blockIdx.x * 256 + threadIdx.x;
  if (i < n) out[i] = (bf16_t)in[i];
}

// ---------------------------------------------------------------------------
// Projection GEMM: C[M=4096, N=1024] = A[M,K=1024] @ W[N,K]^T (bf16, f32 acc)
// A tile (64 rows x 32 k) and W tile (64 cols x 32 k) async-staged in LDS,
// double-buffered, manual unroll-by-2 for compile-time buffer selection.
// mode 0: bf16 Q [B,H,S,D] (scaled)   mode 1: bf16 K [B,H,S,D]
// mode 2: bf16 V^T [B,H,D,S]          mode 3: f32 output [M,N]
// ---------------------------------------------------------------------------
__global__ void __launch_bounds__(128)
gemm_wmma_kernel(const bf16_t* __restrict__ A, const bf16_t* __restrict__ W,
                 bf16_t* __restrict__ outb, float* __restrict__ outf,
                 int mode, float scale) {
  __shared__ bf16_t sA[2][64 * 32];  // [buf][m_local*32 + k_local], 8 KB
  __shared__ bf16_t sB[2][64 * 32];  // [buf][n_local*32 + k_local], 8 KB

  const int tid  = threadIdx.x;
  const int wave = tid >> 5;
  const int lane = tid & 31;
  const int col  = lane & 15;
  const int hi   = lane >> 4;

  const int r0 = blockIdx.x * 64;  // block row base (M)
  const int c0 = blockIdx.y * 64;  // block col base (N)

  v8f acc[4];
#pragma unroll
  for (int t = 0; t < 4; ++t) acc[t] = vzero8();

  // Stage A[r0..r0+63][k0..k0+32) and W[c0..c0+63][k0..k0+32): 2x 256 chunks
  // of 16B, 4 async ops per thread per tile.
  const int crow  = tid >> 2;        // 0..31 -> rows tid/4 and tid/4+32
  const int coff8 = (tid & 3) * 8;   // halves within row
  auto stage_tile = [&](int buf, int k0) {
#pragma unroll
    for (int i = 0; i < 2; ++i) {
      const int row = crow + i * 32;
      const unsigned la = (unsigned)(uintptr_t)&sA[buf][row * 32 + coff8];
      const unsigned lb = (unsigned)(uintptr_t)&sB[buf][row * 32 + coff8];
      async_copy_b128(la, A + (size_t)(r0 + row) * E_ + k0 + coff8);
      async_copy_b128(lb, W + (size_t)(c0 + row) * E_ + k0 + coff8);
    }
  };

  // Consume one staged 32-deep k-slice from buffer `buf` (compile-time).
  auto body = [&](int buf) {
    // A fragment 16x32 from LDS: lanes 0-15 K 0..7 & 16..23; 16-31 K 8..15 & 24..31
    const bf16_t* ap = &sA[buf][(wave * 16 + col) * 32 + hi * 8];
    v16bf af = ld16(ap, ap + 16);
#pragma unroll
    for (int t = 0; t < 4; ++t) {
      // B fragment 32x16 from LDS: lane = column; lanes 0-15 K 0..7 & 16..23; 16-31 K 8..15 & 24..31
      const bf16_t* bp = &sB[buf][(t * 16 + col) * 32 + hi * 8];
      v16bf bfr = ld16(bp, bp + 16);
      acc[t] = wmma_bf16(af, bfr, acc[t]);
    }
  };

  stage_tile(0, 0);
  for (int it = 0; it < E_ / 64; ++it) {
    const int k0 = it * 64;
    // ---- body 0: consume buf0 (k-slice k0), prefetch k0+32 into buf1 ----
    stage_tile(1, k0 + 32);
    wait_async_le4();
    __syncthreads();
    body(0);
    __syncthreads();
    // ---- body 1: consume buf1 (k-slice k0+32), prefetch k0+64 into buf0 ----
    stage_tile(0, (it + 1 < E_ / 64) ? k0 + 64 : k0 + 32);  // clamp: redundant reload
    wait_async_le4();
    __syncthreads();
    body(1);
    __syncthreads();
  }

  // C layout: VGPR r holds row (r0 + wave*16 + r + hi*8), column (c0 + 16t + col)
#pragma unroll
  for (int t = 0; t < 4; ++t) {
#pragma unroll
    for (int r = 0; r < 8; ++r) {
      const int mrow = r0 + wave * 16 + r + hi * 8;
      const int n    = c0 + t * 16 + col;
      const float v  = acc[t][r] * scale;
      const int b = mrow >> 11;
      const int s = mrow & (S_ - 1);
      const int h = n >> 6;
      const int d = n & (D_ - 1);
      if (mode == 0 || mode == 1) {
        outb[(((size_t)b * H_ + h) * S_ + s) * D_ + d] = (bf16_t)v;
      } else if (mode == 2) {
        outb[(((size_t)b * H_ + h) * D_ + d) * S_ + s] = (bf16_t)v;
      } else {
        outf[(size_t)mrow * E_ + n] = v;
      }
    }
  }
}

// ---------------------------------------------------------------------------
// K fragments for keys [k0,k0+32): two 16-key B fragments per 32-d chunk.
// ---------------------------------------------------------------------------
__device__ __forceinline__ void load_kfrag4(const bf16_t* __restrict__ Kp,
                                            int k0, int col, int hi, v16bf kf[4]) {
  const size_t ka = (size_t)(k0 + col) * D_ + (size_t)hi * 8;
  const size_t kb = (size_t)(k0 + 16 + col) * D_ + (size_t)hi * 8;
  kf[0] = ld16(Kp + ka, Kp + ka + 16);            // keys k0..k0+15, d 0..31
  kf[1] = ld16(Kp + kb, Kp + kb + 16);            // keys k0+16..+31, d 0..31
  kf[2] = ld16(Kp + ka + 32, Kp + ka + 48);       // d 32..63
  kf[3] = ld16(Kp + kb + 32, Kp + kb + 48);
}

__device__ __forceinline__ void score_from(const v16bf& qa0, const v16bf& qa1,
                                           const v16bf kf[4], v8f& c0, v8f& c1) {
  c0 = wmma_bf16(qa0, kf[0], c0);
  c1 = wmma_bf16(qa0, kf[1], c1);
  c0 = wmma_bf16(qa1, kf[2], c0);
  c1 = wmma_bf16(qa1, kf[3], c1);
}

// ---------------------------------------------------------------------------
// Flash attention: one wave per 16 query rows; 4 waves per block.
// Pass A: online softmax + O accumulation (V fragments prefetched ahead of
// the softmax VALU block). Pass B: recompute scores (K fragments register-
// pipelined) and write normalized probabilities [B,H,S,S] f32 exactly once.
// ---------------------------------------------------------------------------
__global__ void __launch_bounds__(128)
attn_kernel(const bf16_t* __restrict__ Q, const bf16_t* __restrict__ Kg,
            const bf16_t* __restrict__ Vt, const float* __restrict__ mp,
            float* __restrict__ aw, bf16_t* __restrict__ AO) {
  __shared__ bf16_t sP[4][16 * 32];  // per-wave P staging (C-layout -> A-frag)

  const int wave = threadIdx.x >> 5;
  const int lane = threadIdx.x & 31;
  const int col  = lane & 15;
  const int hi   = lane >> 4;

  const int qb64 = blockIdx.x % (S_ / 64);
  const int bh   = blockIdx.x / (S_ / 64);
  const int h    = bh % H_;
  const int b    = bh / H_;
  const int q_base = qb64 * 64 + wave * 16;
  const int q_max  = q_base + 15;

  const bf16_t* Qp = Q  + (size_t)bh * S_ * D_;
  const bf16_t* Kp = Kg + (size_t)bh * S_ * D_;
  const bf16_t* Vp = Vt + (size_t)bh * D_ * S_;
  const float* mpp = mp + (size_t)h * S_ * S_;
  float* awp       = aw + (size_t)bh * S_ * S_;

  // Q A-fragments (16 rows x d 0..31 / 32..63); Q pre-scaled by 1/sqrt(D)
  const size_t qoff = (size_t)(q_base + col) * D_ + hi * 8;
  v16bf qa0 = ld16(Qp + qoff, Qp + qoff + 16);
  v16bf qa1 = ld16(Qp + qoff + 32, Qp + qoff + 48);

  float m[8], l[8];
  v8f o[4];
#pragma unroll
  for (int r = 0; r < 8; ++r) { m[r] = -INFF; l[r] = 0.f; }
#pragma unroll
  for (int t = 0; t < 4; ++t) o[t] = vzero8();

  // ---------------- Pass A: online softmax + O accumulation ----------------
  for (int k0 = 0; k0 <= q_max; k0 += 32) {
    // V fragments issued first: in flight during the softmax VALU below.
    v16bf vf[4];
#pragma unroll
    for (int t = 0; t < 4; ++t) {
      const size_t vr = (size_t)(t * 16 + col) * S_ + k0 + hi * 8;
      vf[t] = ld16(Vp + vr, Vp + vr + 16);
    }

    v16bf kf[4];
    load_kfrag4(Kp, k0, col, hi, kf);
    v8f c0 = vzero8(), c1 = vzero8();
    score_from(qa0, qa1, kf, c0, c1);

#pragma unroll
    for (int r = 0; r < 8; ++r) {
      const int qrow = q_base + r + hi * 8;
      const int kc0 = k0 + col, kc1 = k0 + 16 + col;
      float s0 = c0[r] + mpp[(size_t)qrow * S_ + kc0];
      float s1 = c1[r] + mpp[(size_t)qrow * S_ + kc1];
      if (kc0 > qrow) s0 = -INFF;
      if (kc1 > qrow) s1 = -INFF;
      float rm = fmaxf(s0, s1);
      rm = fmaxf(rm, __shfl_xor(rm, 1, 32));
      rm = fmaxf(rm, __shfl_xor(rm, 2, 32));
      rm = fmaxf(rm, __shfl_xor(rm, 4, 32));
      rm = fmaxf(rm, __shfl_xor(rm, 8, 32));
      const float mn = fmaxf(m[r], rm);
      const float sc = __expf(m[r] - mn);
      const float p0 = __expf(s0 - mn);
      const float p1 = __expf(s1 - mn);
      float rs = p0 + p1;
      rs += __shfl_xor(rs, 1, 32);
      rs += __shfl_xor(rs, 2, 32);
      rs += __shfl_xor(rs, 4, 32);
      rs += __shfl_xor(rs, 8, 32);
      l[r] = l[r] * sc + rs;
      m[r] = mn;
#pragma unroll
      for (int t = 0; t < 4; ++t) o[t][r] *= sc;
      sP[wave][(r + hi * 8) * 32 + col]      = (bf16_t)p0;
      sP[wave][(r + hi * 8) * 32 + 16 + col] = (bf16_t)p1;
    }
    __builtin_amdgcn_wave_barrier();  // LDS in-order per wave; block reorder only

    const size_t poff = (size_t)col * 32 + hi * 8;
    v16bf pa = ld16(&sP[wave][poff], &sP[wave][poff + 16]);

#pragma unroll
    for (int t = 0; t < 4; ++t) o[t] = wmma_bf16(pa, vf[t], o[t]);
  }

  // Normalize O, store bf16 attn-output in [B, S, E] layout
  float inv[8];
#pragma unroll
  for (int r = 0; r < 8; ++r) inv[r] = 1.0f / l[r];
#pragma unroll
  for (int t = 0; t < 4; ++t) {
#pragma unroll
    for (int r = 0; r < 8; ++r) {
      const int srow = q_base + r + hi * 8;
      const int e    = h * D_ + t * 16 + col;
      AO[((size_t)b * S_ + srow) * E_ + e] = (bf16_t)(o[t][r] * inv[r]);
    }
  }

  // ---------------- Pass B: write normalized probabilities -----------------
  // Valid (at-or-below diagonal) key steps, K fragments register-pipelined.
  {
    v16bf kf[4];
    load_kfrag4(Kp, 0, col, hi, kf);
    for (int k0 = 0; k0 <= q_max; k0 += 32) {
      const int nk0 = (k0 + 32 <= q_max) ? k0 + 32 : k0;
      v16bf kfn[4];
      load_kfrag4(Kp, nk0, col, hi, kfn);  // next step's loads in flight

      v8f c0 = vzero8(), c1 = vzero8();
      score_from(qa0, qa1, kf, c0, c1);
#pragma unroll
      for (int r = 0; r < 8; ++r) {
        const int qrow = q_base + r + hi * 8;
        const int kc0 = k0 + col, kc1 = k0 + 16 + col;
        const float s0 = c0[r] + mpp[(size_t)qrow * S_ + kc0];
        const float s1 = c1[r] + mpp[(size_t)qrow * S_ + kc1];
        const float p0 = (kc0 > qrow) ? 0.f : __expf(s0 - m[r]) * inv[r];
        const float p1 = (kc1 > qrow) ? 0.f : __expf(s1 - m[r]) * inv[r];
        awp[(size_t)qrow * S_ + kc0] = p0;
        awp[(size_t)qrow * S_ + kc1] = p1;
      }
#pragma unroll
      for (int t = 0; t < 4; ++t) kf[t] = kfn[t];
    }
  }
  // Fully-masked steps: pure zero fill.
  for (int k0 = ((q_max + 32) / 32) * 32; k0 < S_; k0 += 32) {
#pragma unroll
    for (int r = 0; r < 8; ++r) {
      const int qrow = q_base + r + hi * 8;
      awp[(size_t)qrow * S_ + k0 + col]      = 0.f;
      awp[(size_t)qrow * S_ + k0 + 16 + col] = 0.f;
    }
  }
}

// ---------------------------------------------------------------------------
// Host launcher. Inputs: x, mask(unused: causal), mask_p, Wq, Wk, Wv, Wo.
// d_out = [output f32 (B*S*E)] ++ [attn_weights f32 (B*H*S*S)].
// Workspace (48 MB): bf16 x, Wq..Wo, Q, K, V^T, attn-out.
// ---------------------------------------------------------------------------
extern "C" void kernel_launch(void* const* d_in, const int* in_sizes, int n_in,
                              void* d_out, int out_size, void* d_ws, size_t ws_size,
                              hipStream_t stream) {
  (void)in_sizes; (void)n_in; (void)out_size; (void)ws_size;
  const float* x  = (const float*)d_in[0];
  const float* mp = (const float*)d_in[2];
  const float* Wq = (const float*)d_in[3];
  const float* Wk = (const float*)d_in[4];
  const float* Wv = (const float*)d_in[5];
  const float* Wo = (const float*)d_in[6];

  float* out  = (float*)d_out;
  float* attw = out + (size_t)B_ * S_ * E_;

  const size_t NX = (size_t)B_ * S_ * E_;  // 4,194,304
  const size_t NW = (size_t)E_ * E_;       // 1,048,576

  bf16_t* ws  = (bf16_t*)d_ws;
  bf16_t* xb  = ws;
  bf16_t* wqb = xb + NX;
  bf16_t* wkb = wqb + NW;
  bf16_t* wvb = wkb + NW;
  bf16_t* wob = wvb + NW;
  bf16_t* Qb  = wob + NW;
  bf16_t* Kb  = Qb + NX;
  bf16_t* Vtb = Kb + NX;
  bf16_t* AOb = Vtb + NX;

  f32_to_bf16_kernel<<<(NX + 255) / 256, 256, 0, stream>>>(x, xb, (int)NX);
  f32_to_bf16_kernel<<<(NW + 255) / 256, 256, 0, stream>>>(Wq, wqb, (int)NW);
  f32_to_bf16_kernel<<<(NW + 255) / 256, 256, 0, stream>>>(Wk, wkb, (int)NW);
  f32_to_bf16_kernel<<<(NW + 255) / 256, 256, 0, stream>>>(Wv, wvb, (int)NW);
  f32_to_bf16_kernel<<<(NW + 255) / 256, 256, 0, stream>>>(Wo, wob, (int)NW);

  dim3 ggrid(B_ * S_ / 64, E_ / 64);
  gemm_wmma_kernel<<<ggrid, 128, 0, stream>>>(xb, wqb, Qb, nullptr, 0, 0.125f);
  gemm_wmma_kernel<<<ggrid, 128, 0, stream>>>(xb, wkb, Kb, nullptr, 1, 1.0f);
  gemm_wmma_kernel<<<ggrid, 128, 0, stream>>>(xb, wvb, Vtb, nullptr, 2, 1.0f);

  attn_kernel<<<B_ * H_ * (S_ / 64), 128, 0, stream>>>(Qb, Kb, Vtb, mp, attw, AOb);

  gemm_wmma_kernel<<<ggrid, 128, 0, stream>>>(AOb, wob, nullptr, out, 3, 1.0f);
}
